// TensorNetRepresentation_25245817765939
// MI455X (gfx1250) — matrix operand order repaired and checked by reading, not compiled, not run on hardware
//
#include <hip/hip_runtime.h>
#include <hip/hip_bf16.h>

// ---------------------------------------------------------------------------
// TensorNet representation for MI455X (gfx1250, wave32, WMMA).
// All GEMMs use v_wmma_f32_16x16x32_bf16 (bf16 in, f32 accum).
// Working set (~37MB) is fully L2-resident (192MB); we avoid the reference's
// ~900MB E*H*3*3 intermediates by scattering only the 10 compact tensor
// components per (edge,h) with global_atomic_add_f32 (unsafeAtomicAdd).
// ACC is node-major interleaved ([node][comp][h], comp stride 256B) so all 10
// atomics per (node,h) share ONE 64-bit address + immediate offsets.
// bf16 operands are pre-packed once (Zb by embed, ACCB by node_feat) so the
// WMMA feed paths are pure coalesced u32 loads with no conversion VALU.
// ---------------------------------------------------------------------------

#define NN    8192
#define EE    131072
#define HH    64
#define NH    (NN*HH)          // 524288
#define ACCS  640              // floats per node in ACC: 10 comps * 64 h

typedef __attribute__((ext_vector_type(16))) __bf16 v16bf;
typedef __attribute__((ext_vector_type(2)))  __bf16 v2bf;
typedef __attribute__((ext_vector_type(8)))  float  v8f;

union BF16Frag { v16bf v; unsigned int u[8]; };

__device__ __forceinline__ unsigned int pack_bf16(float lo, float hi) {
#if __has_builtin(__builtin_amdgcn_cvt_pk_bf16_f32)
  v2bf p = __builtin_amdgcn_cvt_pk_bf16_f32(lo, hi);
  return __builtin_bit_cast(unsigned int, p);
#else
  unsigned int a = __float_as_uint(lo), b = __float_as_uint(hi);
  a = (a + 0x7FFFu + ((a >> 16) & 1u)) >> 16;          // RNE bf16
  b = (b + 0x7FFFu + ((b >> 16) & 1u)) >> 16;
  return (a & 0xFFFFu) | (b << 16);
#endif
}

// A-matrix (16x32 bf16) K position for lane/vgpr used here (contiguous k order, not the ISA table order):
// VGPR v holds K = (lane>=16 ? 16 : 0) + v*2  (pair K,K+1)
__device__ __forceinline__ int a_kpat(int v, int lane) {
  return ((lane >= 16) ? 16 : 0) + (v << 1);
}

__device__ __forceinline__ v8f wmma_bf16(const BF16Frag& a, const BF16Frag& b, v8f c) {
  return __builtin_amdgcn_wmma_f32_16x16x32_bf16(
      /*neg_a=*/false, a.v, /*neg_b=*/false, b.v,
      /*c_mod=*/(short)0, c, /*reuse_a=*/false, /*reuse_b=*/false);
}

// ---------------------------------------------------------------------------
// Kernel 1: pre-swizzle weights into bf16 B-fragment layout.
// B (KxN, 16-bit): lane l -> column N=l%16; K half selected by l>=16;
// VGPR v holds packed (K = base + 2v, K+1).
// SWZ layout (u32):
//   [0,4096)        emb2  : [t:4][c:4][v:8][lane:32]   (K=128 -> 4 chunks)
//   [4096,7168)     p1/2/3: [p:3][t:4][v:8][lane:32]    (K=32  -> 1 chunk)
//   [7168,13312)    wt0/1/2:[w:3][t:4][c:2][v:8][lane:32](K=64 -> 2 chunks)
// ---------------------------------------------------------------------------
__global__ void prep_swz_kernel(const float* __restrict__ emb2_w,
                                const float* __restrict__ p1w,
                                const float* __restrict__ p2w,
                                const float* __restrict__ p3w,
                                const float* __restrict__ wt0,
                                const float* __restrict__ wt1,
                                const float* __restrict__ wt2,
                                unsigned int* __restrict__ swz) {
  int idx = blockIdx.x * blockDim.x + threadIdx.x;
  if (idx < 4096) {                       // emb2_w: (128,64), B[K][N]=emb2[K*64+N]
    int lane = idx & 31, v = (idx >> 5) & 7, c = (idx >> 8) & 3, t = (idx >> 10) & 3;
    int Ncol = t * 16 + (lane & 15);
    int K = c * 32 + ((lane >= 16) ? 16 : 0) + v * 2;
    swz[idx] = pack_bf16(emb2_w[K * 64 + Ncol], emb2_w[(K + 1) * 64 + Ncol]);
  } else if (idx < 7168) {                // p*_w: (64,32), B[K][N]=p[N*32+K]
    int r = idx - 4096;
    int lane = r & 31, v = (r >> 5) & 7, t = (r >> 8) & 3, p = (r >> 10);
    const float* pw = (p == 0) ? p1w : ((p == 1) ? p2w : p3w);
    int Ncol = t * 16 + (lane & 15);
    int K = ((lane >= 16) ? 16 : 0) + v * 2;
    swz[idx] = pack_bf16(pw[Ncol * 32 + K], pw[Ncol * 32 + K + 1]);
  } else if (idx < 13312) {               // wt*: (64,64), B[K][N]=wt[N*64+K]
    int r = idx - 7168;
    int lane = r & 31, v = (r >> 5) & 7, c = (r >> 8) & 1, t = (r >> 9) & 3, w = (r >> 11);
    const float* ww = (w == 0) ? wt0 : ((w == 1) ? wt1 : wt2);
    int Ncol = t * 16 + (lane & 15);
    int K = c * 32 + ((lane >= 16) ? 16 : 0) + v * 2;
    swz[idx] = pack_bf16(ww[Ncol * 64 + K], ww[Ncol * 64 + K + 1]);
  }
}

// ---------------------------------------------------------------------------
// Kernel 2: Zb[n][p] = packed bf16 of emb[atomic_numbers[n]][2p,2p+1].
// Pre-packing here makes every edge-kernel A-fragment dword a single u32 load.
// ---------------------------------------------------------------------------
__global__ void embed_kernel(const int* __restrict__ an,
                             const float* __restrict__ emb,
                             unsigned int* __restrict__ Zb) {
  int i = blockIdx.x * 256 + threadIdx.x;   // NN*32 threads
  int n = i >> 5, p = i & 31;
  const float2* e2 = (const float2*)(emb + an[n] * 64);
  float2 v = e2[p];
  Zb[i] = pack_bf16(v.x, v.y);
}

// ---------------------------------------------------------------------------
// Kernel 3: zero the 10-component f32 node accumulators.
// ---------------------------------------------------------------------------
__global__ void zero_kernel(float4* __restrict__ p) {
  int i = blockIdx.x * 256 + threadIdx.x;   // 10*NH/4 threads
  p[i] = make_float4(0.f, 0.f, 0.f, 0.f);
}

// ---------------------------------------------------------------------------
// Kernel 4: edge kernel. One wave per 16 edges; 8 waves (128 edges) / block.
// Per 16-edge tile:  attr(16x32 bf16) x p*_swz -> f1/f2/f3 (3 WMMA/tile)
//                    Zcat(16x128 bf16) x emb2_swz -> Zij   (4 WMMA/tile)
// Then fold rcut/bias and atomically scatter the 10 compact components
// against a single base address (+256B immediate offsets).
// ---------------------------------------------------------------------------
__global__ __launch_bounds__(256) void edge_kernel(
    const int* __restrict__ edge_index, const float* __restrict__ edge_vec,
    const float* __restrict__ edge_weight, const unsigned int* __restrict__ Zb,
    const unsigned int* __restrict__ swz,
    const float* __restrict__ p1b, const float* __restrict__ p2b,
    const float* __restrict__ p3b, float* __restrict__ acc) {
  __shared__ float srec[8][16][10];  // rcut, vx,vy,vz, sxx,syy,szz,sxy,sxz,syz
  __shared__ int   snode[8][16];
  const int wave = threadIdx.x >> 5, lane = threadIdx.x & 31;
  const int ebase = (blockIdx.x * 8 + wave) * 16;

  if (lane < 16) {
    int e = ebase + lane;
    int i = edge_index[e], j = edge_index[EE + e];
    float d = edge_weight[e];
    float rcut = (d < 5.0f) ? 0.5f * (__cosf(d * 0.6283185307f) + 1.0f) : 0.0f;
    float inv = (i == j) ? 1.0f : (1.0f / d);
    float vx = edge_vec[e * 3 + 0] * inv;
    float vy = edge_vec[e * 3 + 1] * inv;
    float vz = edge_vec[e * 3 + 2] * inv;
    float tr3 = (vx * vx + vy * vy + vz * vz) * (1.0f / 3.0f);
    srec[wave][lane][0] = rcut;
    srec[wave][lane][1] = vx; srec[wave][lane][2] = vy; srec[wave][lane][3] = vz;
    srec[wave][lane][4] = vx * vx - tr3;
    srec[wave][lane][5] = vy * vy - tr3;
    srec[wave][lane][6] = vz * vz - tr3;
    srec[wave][lane][7] = vx * vy;
    srec[wave][lane][8] = vx * vz;
    srec[wave][lane][9] = vy * vz;
    snode[wave][lane] = i;
  }
  __syncthreads();

  const int m16 = lane & 15;
  const int e = ebase + m16;

  // edge_attr fragment (16x32 bf16, rcut folded in as in the reference)
  BF16Frag attrA;
  {
    float d = edge_weight[e];
    float rcut = srec[wave][m16][0];
    float start = __expf(-5.0f);
    float tmp = 0.0625f * (1.0f - start);
    float beta = 1.0f / (tmp * tmp);
    float ed = __expf(-d);
#pragma unroll
    for (int v = 0; v < 8; ++v) {
      int K = a_kpat(v, lane);
      float m0 = start + (1.0f - start) * (K * (1.0f / 31.0f));
      float m1 = start + (1.0f - start) * ((K + 1) * (1.0f / 31.0f));
      float a0 = __expf(-beta * (ed - m0) * (ed - m0)) * rcut;
      float a1 = __expf(-beta * (ed - m1) * (ed - m1)) * rcut;
      attrA.u[v] = pack_bf16(a0, a1);
    }
  }

  // concat(Z[i],Z[j]) fragments (16x128 bf16 -> 4 K-chunks).
  // Zb is pre-packed: each fragment dword is one u32 load (L2-resident).
  BF16Frag zA[4];
  {
    int i = edge_index[e], j = edge_index[EE + e];
    const unsigned int* Zi = Zb + i * 32;
    const unsigned int* Zj = Zb + j * 32;
#pragma unroll
    for (int c = 0; c < 4; ++c)
#pragma unroll
      for (int v = 0; v < 8; ++v) {
        int K = c * 32 + a_kpat(v, lane);
        zA[c].u[v] = (K < 64) ? Zi[K >> 1] : Zj[(K - 64) >> 1];
      }
  }

  const int hiM = (lane >= 16) ? 8 : 0;
#pragma unroll
  for (int t = 0; t < 4; ++t) {
    // Zij tile: 4 chained WMMAs over K=128
    v8f zij = {};
#pragma unroll
    for (int c = 0; c < 4; ++c) {
      BF16Frag b;
      const unsigned int* p = swz + ((t * 4 + c) * 8) * 32 + lane;
#pragma unroll
      for (int v = 0; v < 8; ++v) b.u[v] = p[v * 32];
      zij = wmma_bf16(zA[c], b, zij);
    }
    // f1/f2/f3 tiles: one WMMA each (K=32)
    v8f f1 = {}, f2 = {}, f3 = {};
#pragma unroll
    for (int p = 0; p < 3; ++p) {
      BF16Frag b;
      const unsigned int* q = swz + 4096 + ((p * 4 + t) * 8) * 32 + lane;
#pragma unroll
      for (int v = 0; v < 8; ++v) b.u[v] = q[v * 32];
      if (p == 0) f1 = wmma_bf16(attrA, b, f1);
      else if (p == 1) f2 = wmma_bf16(attrA, b, f2);
      else f3 = wmma_bf16(attrA, b, f3);
    }

    int h = t * 16 + m16;
    float b1 = p1b[h], b2 = p2b[h], b3 = p3b[h];
#pragma unroll
    for (int r = 0; r < 8; ++r) {
      int m = r + hiM;                       // edge slot in tile (D layout)
      float rcut = srec[wave][m][0];
      float vx = srec[wave][m][1], vy = srec[wave][m][2], vz = srec[wave][m][3];
      float sxx = srec[wave][m][4], syy = srec[wave][m][5], szz = srec[wave][m][6];
      float sxy = srec[wave][m][7], sxz = srec[wave][m][8], syz = srec[wave][m][9];
      int node = snode[wave][m];
      float zv = zij[r] * rcut;              // C = rcut * Zij
      float g1 = (f1[r] + b1) * zv;
      float g2 = (f2[r] + b2) * zv;
      float g3 = (f3[r] + b3) * zv;
      // One base address; comp offsets (256B apart) fold into IOFFSET.
      float* A0 = acc + node * ACCS + h;
      unsafeAtomicAdd(A0 + 0 * HH, g1);        // I (diagonal scalar)
      unsafeAtomicAdd(A0 + 1 * HH, g2 * vx);   // skew comps
      unsafeAtomicAdd(A0 + 2 * HH, g2 * vy);
      unsafeAtomicAdd(A0 + 3 * HH, g2 * vz);
      unsafeAtomicAdd(A0 + 4 * HH, g3 * sxx);  // sym comps
      unsafeAtomicAdd(A0 + 5 * HH, g3 * syy);
      unsafeAtomicAdd(A0 + 6 * HH, g3 * szz);
      unsafeAtomicAdd(A0 + 7 * HH, g3 * sxy);
      unsafeAtomicAdd(A0 + 8 * HH, g3 * sxz);
      unsafeAtomicAdd(A0 + 9 * HH, g3 * syz);
    }
  }
}

// ---------------------------------------------------------------------------
// Kernel 5: per-node features. Wave per node (8 nodes/block).
// Lane owns h = 2*lane, 2*lane+1 (consecutive pair) so ACC reads are b64 and
// the bf16 mirror ACCB (comp-major, for node_out A fragments) packs directly.
// tn = 3*s1^2 + 2*|a|^2 + ||sym||_F^2 (I/A/S cross terms vanish), then
// LayerNorm -> silu(x@ws1.T+bs1) -> silu(@ws2.T+bs2) -> norm (N,64,3).
// ---------------------------------------------------------------------------
__global__ __launch_bounds__(256) void node_feat_kernel(
    const float* __restrict__ acc, const float* __restrict__ ws1,
    const float* __restrict__ bs1, const float* __restrict__ ws2,
    const float* __restrict__ bs2, const float* __restrict__ ln_g,
    const float* __restrict__ ln_b, unsigned int* __restrict__ accB,
    float* __restrict__ normout) {
  __shared__ float sx[8][64];
  __shared__ float sy[8][128];
  int wave = threadIdx.x >> 5, lane = threadIdx.x & 31;
  int n = blockIdx.x * 8 + wave;

  float2 cp[10];
#pragma unroll
  for (int c = 0; c < 10; ++c) {
    const float2* a2 = (const float2*)(acc + n * ACCS + c * HH);
    cp[c] = a2[lane];
    accB[c * (NH / 2) + n * 32 + lane] = pack_bf16(cp[c].x, cp[c].y);
  }
  float tn0 = 3.f * cp[0].x * cp[0].x +
              2.f * (cp[1].x * cp[1].x + cp[2].x * cp[2].x + cp[3].x * cp[3].x) +
              cp[4].x * cp[4].x + cp[5].x * cp[5].x + cp[6].x * cp[6].x +
              2.f * (cp[7].x * cp[7].x + cp[8].x * cp[8].x + cp[9].x * cp[9].x);
  float tn1 = 3.f * cp[0].y * cp[0].y +
              2.f * (cp[1].y * cp[1].y + cp[2].y * cp[2].y + cp[3].y * cp[3].y) +
              cp[4].y * cp[4].y + cp[5].y * cp[5].y + cp[6].y * cp[6].y +
              2.f * (cp[7].y * cp[7].y + cp[8].y * cp[8].y + cp[9].y * cp[9].y);

  float s = tn0 + tn1;
  float ss = tn0 * tn0 + tn1 * tn1;
#pragma unroll
  for (int off = 16; off; off >>= 1) {
    s += __shfl_xor(s, off, 32);
    ss += __shfl_xor(ss, off, 32);
  }
  float mu = s * (1.0f / 64.0f);
  float var = ss * (1.0f / 64.0f) - mu * mu;
  float rstd = rsqrtf(var + 1e-5f);
  int h0 = 2 * lane;
  sx[wave][h0]     = (tn0 - mu) * rstd * ln_g[h0]     + ln_b[h0];
  sx[wave][h0 + 1] = (tn1 - mu) * rstd * ln_g[h0 + 1] + ln_b[h0 + 1];
  __syncthreads();
  for (int k = lane; k < 128; k += 32) {
    float a1 = bs1[k];
    const float* w = ws1 + k * 64;
    for (int h = 0; h < 64; ++h) a1 += sx[wave][h] * w[h];
    sy[wave][k] = a1 / (1.0f + __expf(-a1));   // silu
  }
  __syncthreads();
  for (int m = lane; m < 192; m += 32) {
    float a2 = bs2[m];
    const float* w = ws2 + m * 128;
    for (int k = 0; k < 128; ++k) a2 += sy[wave][k] * w[k];
    normout[n * 192 + m] = a2 / (1.0f + __expf(-a2));
  }
}

// ---------------------------------------------------------------------------
// Kernel 6: node output. Wave per 16-node tile (8 tiles/block).
// 10 GEMMs (16x64 bf16)x(64x16 bf16) per g-tile via chained WMMA; A fragments
// are plain u32 loads from the pre-packed ACCB mirror (no conversion VALU).
// Fused assembly of out[n,g,3,3] with norm gates, written straight to d_out.
// ---------------------------------------------------------------------------
__global__ __launch_bounds__(256) void node_out_kernel(
    const unsigned int* __restrict__ accB, const unsigned int* __restrict__ swz,
    const float* __restrict__ normv, float* __restrict__ out) {
  int wave = threadIdx.x >> 5, lane = threadIdx.x & 31;
  int tileN = blockIdx.x * 8 + wave;
  int nbase = tileN * 16;
  int m16 = lane & 15;
  int hiM = (lane >= 16) ? 8 : 0;
  const unsigned int* swzWt = swz + 7168;

#pragma unroll
  for (int t = 0; t < 4; ++t) {
    v8f u[10];
#pragma unroll
    for (int comp = 0; comp < 10; ++comp) {
      int w = (comp == 0) ? 0 : ((comp < 4) ? 1 : 2);   // wt0 / wt1 / wt2
      const unsigned int* src = accB + comp * (NH / 2) + (nbase + m16) * 32;
      v8f a = {};
#pragma unroll
      for (int c = 0; c < 2; ++c) {
        BF16Frag A;
#pragma unroll
        for (int v = 0; v < 8; ++v) {
          int K = c * 32 + a_kpat(v, lane);
          A.u[v] = src[K >> 1];
        }
        BF16Frag B;
        const unsigned int* p = swzWt + (((w * 4 + t) * 2 + c) * 8) * 32 + lane;
#pragma unroll
        for (int v = 0; v < 8; ++v) B.u[v] = p[v * 32];
        a = wmma_bf16(A, B, a);
      }
      u[comp] = a;
    }
    int g = t * 16 + m16;
#pragma unroll
    for (int r = 0; r < 8; ++r) {
      int n = nbase + r + hiM;
      const float* nm = normv + n * 192 + g * 3;
      float n0 = nm[0], n1 = nm[1], n2 = nm[2];
      float i0 = u[0][r] * n0;
      float wx = u[1][r] * n1, wy = u[2][r] * n1, wz = u[3][r] * n1;
      float mxx = u[4][r] * n2, myy = u[5][r] * n2, mzz = u[6][r] * n2;
      float mxy = u[7][r] * n2, mxz = u[8][r] * n2, myz = u[9][r] * n2;
      float* o = out + (n * 64 + g) * 9;
      o[0] = i0 + mxx;  o[1] = -wz + mxy; o[2] =  wy + mxz;
      o[3] =  wz + mxy; o[4] = i0 + myy;  o[5] = -wx + myz;
      o[6] = -wy + mxz; o[7] =  wx + myz; o[8] = i0 + mzz;
    }
  }
}

// ---------------------------------------------------------------------------
extern "C" void kernel_launch(void* const* d_in, const int* in_sizes, int n_in,
                              void* d_out, int out_size, void* d_ws, size_t ws_size,
                              hipStream_t stream) {
  (void)in_sizes; (void)n_in; (void)out_size; (void)ws_size;
  const int*   atomic_numbers = (const int*)d_in[0];
  const int*   edge_index     = (const int*)d_in[1];
  const float* edge_vec       = (const float*)d_in[2];
  const float* edge_weight    = (const float*)d_in[3];
  const float* emb            = (const float*)d_in[4];
  const float* emb2_w         = (const float*)d_in[5];
  const float* p1_w = (const float*)d_in[6];  const float* p1_b = (const float*)d_in[7];
  const float* p2_w = (const float*)d_in[8];  const float* p2_b = (const float*)d_in[9];
  const float* p3_w = (const float*)d_in[10]; const float* p3_b = (const float*)d_in[11];
  const float* wt0 = (const float*)d_in[12];
  const float* wt1 = (const float*)d_in[13];
  const float* wt2 = (const float*)d_in[14];
  const float* ws1 = (const float*)d_in[15]; const float* bs1 = (const float*)d_in[16];
  const float* ws2 = (const float*)d_in[17]; const float* bs2 = (const float*)d_in[18];
  const float* ln_g = (const float*)d_in[19]; const float* ln_b = (const float*)d_in[20];
  float* out = (float*)d_out;

  // Workspace layout (total ~37.1 MB):
  char* ws = (char*)d_ws;
  unsigned int* ZB   = (unsigned int*)(ws);                        //  1 MB
  float*        ACC  = (float*)(ws + (size_t)(1u << 20));          // 20 MB f32
  unsigned int* ACCB = (unsigned int*)(ws + (size_t)(21u << 20));  // 10 MB bf16
  float*        NORM = (float*)(ws + (size_t)(31u << 20));         //  6 MB
  unsigned int* SWZ  = (unsigned int*)(ws + (size_t)(37u << 20));  // 52 KB

  prep_swz_kernel<<<52, 256, 0, stream>>>(emb2_w, p1_w, p2_w, p3_w, wt0, wt1, wt2, SWZ);
  embed_kernel<<<(NN * 32) / 256, 256, 0, stream>>>(atomic_numbers, emb, ZB);
  zero_kernel<<<(10 * NH / 4) / 256, 256, 0, stream>>>((float4*)ACC);
  edge_kernel<<<EE / 128, 256, 0, stream>>>(edge_index, edge_vec, edge_weight, ZB,
                                            SWZ, p1_b, p2_b, p3_b, ACC);
  node_feat_kernel<<<NN / 8, 256, 0, stream>>>(ACC, ws1, bs1, ws2, bs2, ln_g, ln_b,
                                               ACCB, NORM);
  node_out_kernel<<<NN / 128, 256, 0, stream>>>(ACCB, SWZ, NORM, out);
}
